// Playlist_Model_89275190215119
// MI455X (gfx1250) — matrix operand order repaired and checked by reading, not compiled, not run on hardware
//
#include <hip/hip_runtime.h>
#include <stdint.h>

// ---------------------------------------------------------------------------
// CDNA5 / gfx1250 WMMA types (wave32). bf16 storage is kept as unsigned short
// everywhere; only the WMMA fragments use the __bf16 ext-vector types.
// ---------------------------------------------------------------------------
typedef __attribute__((ext_vector_type(8)))  __bf16 v8bf;
typedef __attribute__((ext_vector_type(16))) __bf16 v16bf;
typedef __attribute__((ext_vector_type(8)))  float  v8f;

__device__ __forceinline__ unsigned short f2bfbits(float f) {
  union { float f; unsigned int u; } c; c.f = f;
  unsigned int u = c.u;
  u += 0x7FFFu + ((u >> 16) & 1u);          // round-to-nearest-even
  return (unsigned short)(u >> 16);
}

struct PoolArgs {
  const float* tbl[12];
  const int*   idx[12];
  int          len[12];
};

struct __align__(8) U2 { unsigned int x, y; };

// ---------------------------------------------------------------------------
// Kernel 1: embedding-bag mean pool.
// One block per (batch b, feature f). 128 threads x float4 = 512 cols.
// Accumulate fp32 over L rows, write bf16 into X[b, f*512 .. f*512+511].
// ---------------------------------------------------------------------------
__global__ __launch_bounds__(128)
void pool_embed_kernel(PoolArgs args, unsigned short* __restrict__ Xb) {
  __shared__ int sidx[128];
  const int b = blockIdx.x;
  const int f = blockIdx.y;
  const int t = threadIdx.x;
  const int Lf = args.len[f];
  if (t < Lf) sidx[t] = args.idx[f][(size_t)b * Lf + t];
  __syncthreads();
  const float* T = args.tbl[f];
  float ax = 0.f, ay = 0.f, az = 0.f, aw = 0.f;
  for (int l = 0; l < Lf; ++l) {
    const float4 e = *(const float4*)(T + (size_t)sidx[l] * 512 + t * 4);
    ax += e.x; ay += e.y; az += e.z; aw += e.w;
  }
  const float inv = 1.0f / (float)Lf;
  U2 p;
  p.x = (unsigned)f2bfbits(ax * inv) | ((unsigned)f2bfbits(ay * inv) << 16);
  p.y = (unsigned)f2bfbits(az * inv) | ((unsigned)f2bfbits(aw * inv) << 16);
  *(U2*)(Xb + (size_t)b * 6144 + (size_t)f * 512 + (size_t)t * 4) = p;
}

// ---------------------------------------------------------------------------
// Kernel 2: W [K,N] fp32 row-major  ->  Wt [N,K] bf16 ("column-major" W).
// With Wt N-major, each lane's WMMA B-fragment is one contiguous 32B load.
// ---------------------------------------------------------------------------
__global__ __launch_bounds__(256)
void wconvert_kernel(const float* __restrict__ W, unsigned short* __restrict__ Wt,
                     int K, int N) {
  size_t id = (size_t)blockIdx.x * blockDim.x + threadIdx.x;
  if (id >= (size_t)K * N) return;
  int n = (int)(id / (size_t)K);
  int k = (int)(id % (size_t)K);
  Wt[id] = f2bfbits(W[(size_t)k * N + n]);
}

// ---------------------------------------------------------------------------
// Kernel 3: O = [relu](A @ W + bias) via v_wmma_f32_16x16x32_bf16.
// A:  bf16 row-major [M,K].  Wt: bf16 [N,K].  bias: fp32 [N].
// One wave computes a 16x64 strip: A-fragment reused across 4 B-fragments.
// The loop body is split into a load-phase (distinct registers for A and all
// four B fragments) and a WMMA-phase, so the scheduler can clause the loads,
// retire WMMAs against partial s_wait_loadcnt, and overlap the next
// iteration's loads with the current iteration's matrix ops.
// Fragment layouts used here (wave32, same contiguous k order for A and B):
//   A 16x32 bf16 : lane(l16,hv) row=l16, K = hv*16+{0..15}
//                  -> two contiguous 16B loads from row-major A.
//   B 32x16 bf16 : lane col=l16, K = hv*16+{0..15}
//                  -> one contiguous 32B load from N-major Wt.
//   C/D 16x16 f32: VGPR r holds rows r (lanes 0-15) and r+8 (lanes 16-31).
// ---------------------------------------------------------------------------
__global__ __launch_bounds__(256)
void gemm_bf16_wmma_kernel(const unsigned short* __restrict__ A,
                           const unsigned short* __restrict__ Wt,
                           const float* __restrict__ bias,
                           float* __restrict__ outF,            // fp32 out (or null)
                           unsigned short* __restrict__ outB,   // bf16 out (or null)
                           int M, int K, int N, int relu) {
  const int wid  = (int)((blockIdx.x * blockDim.x + threadIdx.x) >> 5);
  const int lane = (int)(threadIdx.x & 31);
  const int numMt = M >> 4;
  const int numN4 = N >> 6;
  if (wid >= numMt * numN4) return;            // wave-uniform: EXEC stays all-1s
  const int mt = wid % numMt;
  const int n4 = wid / numMt;
  const int hv  = lane >> 4;                   // half-wave select
  const int l16 = lane & 15;

  const unsigned short* arow = A + (size_t)((mt << 4) + l16) * K + (hv << 4);
  const int ncol = (n4 << 6) + l16;

  const unsigned short* bp0 = Wt + (size_t)(ncol)      * K + (hv << 4);
  const unsigned short* bp1 = Wt + (size_t)(ncol + 16) * K + (hv << 4);
  const unsigned short* bp2 = Wt + (size_t)(ncol + 32) * K + (hv << 4);
  const unsigned short* bp3 = Wt + (size_t)(ncol + 48) * K + (hv << 4);

  v8f acc0 = {}, acc1 = {}, acc2 = {}, acc3 = {};

#pragma unroll 2
  for (int k0 = 0; k0 < K; k0 += 32) {
    // ---- load phase: all fragments into distinct registers ----
    union { v16bf v; v8bf h[2]; } a;
    a.h[0] = *(const v8bf*)(arow + k0);        // K = k0 + hv*16 + 0..7
    a.h[1] = *(const v8bf*)(arow + k0 + 8);    // K = k0 + hv*16 + 8..15
    v16bf bf0 = *(const v16bf*)(bp0 + k0);
    v16bf bf1 = *(const v16bf*)(bp1 + k0);
    v16bf bf2 = *(const v16bf*)(bp2 + k0);
    v16bf bf3 = *(const v16bf*)(bp3 + k0);
    // ---- compute phase: 4 WMMAs sharing the A fragment ----
    acc0 = __builtin_amdgcn_wmma_f32_16x16x32_bf16(false, a.v, false, bf0,
                                                   (short)0, acc0, false, false);
    acc1 = __builtin_amdgcn_wmma_f32_16x16x32_bf16(false, a.v, false, bf1,
                                                   (short)0, acc1, false, false);
    acc2 = __builtin_amdgcn_wmma_f32_16x16x32_bf16(false, a.v, false, bf2,
                                                   (short)0, acc2, false, false);
    acc3 = __builtin_amdgcn_wmma_f32_16x16x32_bf16(false, a.v, false, bf3,
                                                   (short)0, acc3, false, false);
  }

  // Epilogue: bias + optional relu, store fp32 or bf16 row-major [M,N].
  v8f accs[4] = { acc0, acc1, acc2, acc3 };
#pragma unroll
  for (int j = 0; j < 4; ++j) {
    const int col = ncol + 16 * j;
    const float bv = bias[col];
#pragma unroll
    for (int r = 0; r < 8; ++r) {
      const int row = (mt << 4) + (hv << 3) + r;
      float v = accs[j][r] + bv;
      if (relu) v = v > 0.f ? v : 0.f;
      if (outF) outF[(size_t)row * N + col] = v;
      else      outB[(size_t)row * N + col] = f2bfbits(v);
    }
  }
}

// ---------------------------------------------------------------------------
// Launch: pool -> (3x weight convert) -> 3x WMMA GEMM. All on `stream`.
// ---------------------------------------------------------------------------
extern "C" void kernel_launch(void* const* d_in, const int* in_sizes, int n_in,
                              void* d_out, int out_size, void* d_ws, size_t ws_size,
                              hipStream_t stream) {
  (void)in_sizes; (void)n_in; (void)out_size; (void)ws_size;
  const int B = 1024, D = 512;
  const int K1 = 12 * D, N1 = 512, N2 = 256, N3 = 128;

  PoolArgs pa;
  for (int i = 0; i < 12; ++i) pa.tbl[i] = (const float*)d_in[i];
  // features 0-2: single-id towers; 3-11: L=100 sequence towers
  for (int i = 0; i < 3; ++i) { pa.idx[i] = (const int*)d_in[18 + i]; pa.len[i] = 1; }
  for (int i = 0; i < 9; ++i) { pa.idx[3 + i] = (const int*)d_in[21 + i]; pa.len[3 + i] = 100; }

  const float* W1 = (const float*)d_in[12]; const float* b1 = (const float*)d_in[13];
  const float* W2 = (const float*)d_in[14]; const float* b2 = (const float*)d_in[15];
  const float* W3 = (const float*)d_in[16]; const float* b3 = (const float*)d_in[17];

  // Workspace layout (bf16 elements), ~20.8 MB total.
  unsigned short* ws = (unsigned short*)d_ws;
  size_t o = 0;
  unsigned short* Xb  = ws + o; o += (size_t)B  * K1;   // 1024 x 6144
  unsigned short* W1t = ws + o; o += (size_t)K1 * N1;   // [512, 6144]
  unsigned short* W2t = ws + o; o += (size_t)N1 * N2;   // [256, 512]
  unsigned short* W3t = ws + o; o += (size_t)N2 * N3;   // [128, 256]
  unsigned short* H1  = ws + o; o += (size_t)B  * N1;   // 1024 x 512
  unsigned short* H2  = ws + o; o += (size_t)B  * N2;   // 1024 x 256

  pool_embed_kernel<<<dim3(B, 12), 128, 0, stream>>>(pa, Xb);

  auto cvb = [](size_t n) { return (unsigned)((n + 255) / 256); };
  wconvert_kernel<<<cvb((size_t)K1 * N1), 256, 0, stream>>>(W1, W1t, K1, N1);
  wconvert_kernel<<<cvb((size_t)N1 * N2), 256, 0, stream>>>(W2, W2t, N1, N2);
  wconvert_kernel<<<cvb((size_t)N2 * N3), 256, 0, stream>>>(W3, W3t, N2, N3);

  auto gb = [](int M, int N) { return (unsigned)(((M >> 4) * (N >> 6) + 7) / 8); };
  gemm_bf16_wmma_kernel<<<gb(B, N1), 256, 0, stream>>>(Xb, W1t, b1, nullptr, H1, B, K1, N1, 1);
  gemm_bf16_wmma_kernel<<<gb(B, N2), 256, 0, stream>>>(H1, W2t, b2, nullptr, H2, B, N1, N2, 1);
  gemm_bf16_wmma_kernel<<<gb(B, N3), 256, 0, stream>>>(H2, W3t, b3, (float*)d_out, nullptr, B, N2, N3, 0);
}
